// GAT_5171140625163
// MI455X (gfx1250) — matrix operand order repaired and checked by reading, not compiled, not run on hardware
//
#include <hip/hip_runtime.h>
#include <hip/hip_bf16.h>

// GAT on gfx1250: WMMA (16x16x32 f16->f32) for feature transforms and for the
// flash-attention-style alpha@h aggregation. wave32; A and B fragments use contiguous k order (k = 16*(lane/16) + t).

typedef __attribute__((ext_vector_type(16))) _Float16 v16h;
typedef __attribute__((ext_vector_type(8)))  float    v8f;

#define NB      8
#define NNODES  1024

union AFrag { v16h h; uint4 u[2]; };

// ---------------------------------------------------------------- convert
__global__ void k_f32_to_f16(const float* __restrict__ src,
                             _Float16* __restrict__ dst, int n) {
    int i = blockIdx.x * blockDim.x + threadIdx.x;
    if (i < n) dst[i] = (_Float16)src[i];
}

// ---------------------------------------------------------------- transform
// h[m, f] = sum_k x16[m, k] * w16[f, k]     (M = 8192 rows)
// Writes: hout (f32 row-major, for score kernel) and hB (f16, WMMA-B fragment
// layout: hB[((jblk*nf + ftile)*32 + lane)*16 + t], lane = (k32/16)*16 + n,
// t = k32%16 -> attention B loads are one contiguous 32B read per lane).
__global__ void __launch_bounds__(128) k_transform(
    const _Float16* __restrict__ x16, const _Float16* __restrict__ w16,
    float* __restrict__ hout, _Float16* __restrict__ hB, int K, int Fout)
{
    const int lane = threadIdx.x & 31;
    const int wave = threadIdx.x >> 5;
    const int m0   = blockIdx.x * 64 + wave * 16;
    const int f0   = blockIdx.y * 64;
    const int g    = lane >> 4;
    const int r    = lane & 15;
    const int nf   = Fout >> 4;

    v8f acc[4];
    for (int ct = 0; ct < 4; ++ct)
        for (int e = 0; e < 8; ++e) acc[ct][e] = 0.0f;

    const _Float16* arow = x16 + (size_t)(m0 + r) * K;
    for (int kb = 0; kb < K; kb += 32) {
        // A fragment: row r, halves k = kb + g*16 + t: two 16B runs
        AFrag a;
        a.u[0] = *(const uint4*)(arow + kb + g * 16);
        a.u[1] = *(const uint4*)(arow + kb + g * 16 + 8);
        for (int ct = 0; ct < 4; ++ct) {
            // B fragment: column n=r of B == row (f0+ct*16+r) of W, K-contiguous 16
            const _Float16* wp =
                w16 + (size_t)(f0 + ct * 16 + r) * K + kb + g * 16;
            v16h bf = *(const v16h*)wp;
            acc[ct] = __builtin_amdgcn_wmma_f32_16x16x32_f16(
                false, a.h, false, bf, (short)0, acc[ct], false, false);
        }
    }

    const int jblk  = m0 >> 5;     // constant over the 16-row tile
    const int kbase = m0 & 31;     // 0 or 16
    for (int ct = 0; ct < 4; ++ct) {
        const int f = f0 + ct * 16 + r;          // D column = lane%16 = r
        for (int v = 0; v < 8; ++v) {
            const int m   = v + 8 * g;           // D row
            const float x = acc[ct][v];
            hout[(size_t)(m0 + m) * Fout + f] = x;
            const int k32 = kbase + m;
            const int gb = k32 >> 4, tb = k32 & 15;
            const size_t idx =
                (((size_t)(jblk * nf + (f >> 4)) * 32) + gb * 16 + r) * 16 + tb;
            hB[idx] = (_Float16)x;
        }
    }
}

// ---------------------------------------------------------------- scores
__global__ void k_scores(const float* __restrict__ h,
                         const float* __restrict__ asrc,
                         const float* __restrict__ adst,
                         float* __restrict__ s, float* __restrict__ d,
                         int H, int Fout)
{
    int i = blockIdx.x * blockDim.x + threadIdx.x;     // i = (b*H + hh)*N + n
    if (i >= NB * H * NNODES) return;
    int n  = i % NNODES;
    int bh = i / NNODES;
    int hh = bh % H, b = bh / H;
    const float* row = h + (size_t)(b * NNODES + n) * Fout + hh * 64;
    const float* av  = asrc + hh * 64;
    const float* dv  = adst + hh * 64;
    float ss = 0.0f, dd = 0.0f;
    for (int c = 0; c < 64; ++c) { float x = row[c]; ss += x * av[c]; dd += x * dv[c]; }
    s[i] = ss;
    d[i] = dd;
}

// ---------------------------------------------------------------- attention
// One block = one 16-row i-tile, one wave per head. Flash style: P tiles
// (16x32) built in registers -> WMMA A operand; value fragments streamed from
// hB; f32 accumulation; normalize by running row-sum; +bias, relu, store.
__global__ void __launch_bounds__(128) k_gat_attn(
    const int* __restrict__ adj, const float* __restrict__ sbuf,
    const float* __restrict__ dbuf, const _Float16* __restrict__ hB,
    const float* __restrict__ bias, _Float16* __restrict__ xo16,
    float* __restrict__ xo32, int H, int Fout, int write_f32)
{
    __shared__ int lds_adj[16][33];                  // pad to dodge bank conflicts
    const int lane = threadIdx.x & 31;
    const int w    = threadIdx.x >> 5;               // head
    const int b    = blockIdx.y;
    const int i0   = blockIdx.x * 16;
    const int g    = lane >> 4;
    const int r    = lane & 15;
    const int nf   = Fout >> 4;
    const int nth  = H * 32;

    const float dv = dbuf[((size_t)b * H + w) * NNODES + i0 + r];

    v8f acc[4];
    for (int ct = 0; ct < 4; ++ct)
        for (int e = 0; e < 8; ++e) acc[ct][e] = 0.0f;
    float rowsum = 0.0f;

    for (int j0 = 0; j0 < NNODES; j0 += 32) {
        __syncthreads();
        for (int e = threadIdx.x; e < 128; e += nth) {   // 16x32 ints, uint4 loads
            int rr = e >> 3, c4 = (e & 7) * 4;
            uint4 v4 = *(const uint4*)(adj +
                ((size_t)b * NNODES + i0 + rr) * NNODES + j0 + c4);
            lds_adj[rr][c4 + 0] = v4.x; lds_adj[rr][c4 + 1] = v4.y;
            lds_adj[rr][c4 + 2] = v4.z; lds_adj[rr][c4 + 3] = v4.w;
        }
        if (j0 + 32 < NNODES)                            // global_prefetch_b8 path
            __builtin_prefetch(adj + ((size_t)b * NNODES + i0 + r) * NNODES + j0 + 32, 0, 1);
        __syncthreads();

        const float sv = sbuf[((size_t)b * H + w) * NNODES + j0 + lane];
        AFrag a;
        for (int t = 0; t < 16; ++t) {
            const int k  = g * 16 + t;                       // A-layout k index
            const float sj = __shfl(sv, k, 32);
            float lg = sj + dv;
            lg = lg > 0.0f ? lg : 0.2f * lg;                 // leaky_relu(0.2)
            const bool conn = (lds_adj[r][k] > 0) || (i0 + r == j0 + k);
            const float p = conn ? __expf(lg) : 0.0f;        // logits tiny: safe
            rowsum += p;
            a.h[t] = (_Float16)p;
        }

        const int jblk = (b * NNODES + j0) >> 5;
        const _Float16* hb = hB + ((size_t)(jblk * nf + w * 4) * 32 + lane) * 16;
        for (int ct = 0; ct < 4; ++ct) {
            v16h bf = *(const v16h*)(hb + (size_t)ct * 512);
            acc[ct] = __builtin_amdgcn_wmma_f32_16x16x32_f16(
                false, a.h, false, bf, (short)0, acc[ct], false, false);
        }
    }

    // full row sums live in lanes 0..15 after pairing with lane^16
    const float rs = rowsum + __shfl(rowsum, lane ^ 16, 32);
    float inv[8];
    for (int v = 0; v < 8; ++v) inv[v] = 1.0f / __shfl(rs, v + 8 * g, 32);

    for (int ct = 0; ct < 4; ++ct) {
        const int f  = w * 64 + ct * 16 + r;
        const float bs = bias[f];
        for (int v = 0; v < 8; ++v) {
            const int m = v + 8 * g;
            float val = acc[ct][v] * inv[v] + bs;
            val = val > 0.0f ? val : 0.0f;                   // relu
            const size_t o = ((size_t)b * NNODES + i0 + m) * Fout + f;
            xo16[o] = (_Float16)val;
            if (write_f32) xo32[o] = val;
        }
    }
}

// ---------------------------------------------------------------- readout
__global__ void k_readout_g(const float* __restrict__ x3,
                            const float* __restrict__ wg,
                            const float* __restrict__ bg,
                            const float* __restrict__ wv,
                            float* __restrict__ pv)
{
    __shared__ float gsh[64];
    __shared__ float ggsh[64];
    const int b = blockIdx.x, o = threadIdx.x;
    float s = 0.0f;
    const float* base = x3 + (size_t)b * NNODES * 64 + o;
    for (int n = 0; n < NNODES; ++n) s += base[(size_t)n * 64];
    gsh[o] = s;
    __syncthreads();
    float t = bg[o];
    for (int c = 0; c < 64; ++c) t += gsh[c] * wg[o * 64 + c];
    t = t > 0.0f ? t : 0.0f;
    ggsh[o] = t * wv[64 + o];
    __syncthreads();
    if (o == 0) {
        float p = 0.0f;
        for (int c = 0; c < 64; ++c) p += ggsh[c];
        pv[b] = p;
    }
}

__global__ void k_final(const float* __restrict__ x3, const float* __restrict__ wn,
                        const float* __restrict__ bn, const float* __restrict__ wv,
                        const float* __restrict__ bv, const float* __restrict__ pv,
                        float* __restrict__ out)
{
    int i = blockIdx.x * blockDim.x + threadIdx.x;
    if (i >= NB * NNODES) return;
    const int b = i / NNODES;
    const float* xr = x3 + (size_t)i * 64;
    float xl[64];
    for (int c = 0; c < 64; ++c) xl[c] = xr[c];
    float acc = bv[0] + pv[b];
    for (int o = 0; o < 64; ++o) {
        float t = bn[o];
        const float* wr = wn + o * 64;
        for (int c = 0; c < 64; ++c) t += xl[c] * wr[c];
        t = t > 0.0f ? t : 0.0f;
        acc += wv[o] * t;
    }
    out[i] = acc;
}

// ---------------------------------------------------------------- launch
extern "C" void kernel_launch(void* const* d_in, const int* in_sizes, int n_in,
                              void* d_out, int out_size, void* d_ws, size_t ws_size,
                              hipStream_t stream)
{
    const float* nodef = (const float*)d_in[0];
    const int*   adj   = (const int*)  d_in[1];
    const float* w1 = (const float*)d_in[2],  *as1 = (const float*)d_in[3];
    const float* ad1= (const float*)d_in[4],  *b1  = (const float*)d_in[5];
    const float* w2 = (const float*)d_in[6],  *as2 = (const float*)d_in[7];
    const float* ad2= (const float*)d_in[8],  *b2  = (const float*)d_in[9];
    const float* w3 = (const float*)d_in[10], *as3 = (const float*)d_in[11];
    const float* ad3= (const float*)d_in[12], *b3  = (const float*)d_in[13];
    const float* wn = (const float*)d_in[14], *bn  = (const float*)d_in[15];
    const float* wg = (const float*)d_in[16], *bg  = (const float*)d_in[17];
    const float* wv = (const float*)d_in[18], *bv  = (const float*)d_in[19];

    char* ws = (char*)d_ws;
    constexpr size_t OFF_X16 = 0;                 // 8192*256 f16  = 4 MB
    constexpr size_t OFF_W16 = 4194304;           // 256*256 f16   = 128 KB
    constexpr size_t OFF_H32 = 4325376;           // 8192*256 f32  = 8 MB
    constexpr size_t OFF_HB  = 12713984;          // 8192*256 f16  = 4 MB
    constexpr size_t OFF_S   = 16908288;          // 32768 f32
    constexpr size_t OFF_D   = 17039360;          // 32768 f32
    constexpr size_t OFF_XN  = 17170432;          // 8192*256 f32  = 8 MB
    constexpr size_t OFF_PV  = 25559040;          // 8 f32

    _Float16* x16 = (_Float16*)(ws + OFF_X16);
    _Float16* w16 = (_Float16*)(ws + OFF_W16);
    float*    h32 = (float*)   (ws + OFF_H32);
    _Float16* hB  = (_Float16*)(ws + OFF_HB);
    float*    sb  = (float*)   (ws + OFF_S);
    float*    db  = (float*)   (ws + OFF_D);
    float*    xn  = (float*)   (ws + OFF_XN);
    float*    pv  = (float*)   (ws + OFF_PV);

    const int M = NB * NNODES;                    // 8192

    // ---- layer 1 (K=64, Fout=256, H=4)
    k_f32_to_f16<<<(M * 64 + 255) / 256, 256, 0, stream>>>(nodef, x16, M * 64);
    k_f32_to_f16<<<(256 * 64 + 255) / 256, 256, 0, stream>>>(w1, w16, 256 * 64);
    k_transform<<<dim3(M / 64, 4), 128, 0, stream>>>(x16, w16, h32, hB, 64, 256);
    k_scores<<<(NB * 4 * NNODES + 255) / 256, 256, 0, stream>>>(h32, as1, ad1, sb, db, 4, 256);
    k_gat_attn<<<dim3(NNODES / 16, NB), 128, 0, stream>>>(adj, sb, db, hB, b1, x16, xn, 4, 256, 0);

    // ---- layer 2 (K=256, Fout=256, H=4)
    k_f32_to_f16<<<(256 * 256 + 255) / 256, 256, 0, stream>>>(w2, w16, 256 * 256);
    k_transform<<<dim3(M / 64, 4), 128, 0, stream>>>(x16, w16, h32, hB, 256, 256);
    k_scores<<<(NB * 4 * NNODES + 255) / 256, 256, 0, stream>>>(h32, as2, ad2, sb, db, 4, 256);
    k_gat_attn<<<dim3(NNODES / 16, NB), 128, 0, stream>>>(adj, sb, db, hB, b2, x16, xn, 4, 256, 0);

    // ---- layer 3 (K=256, Fout=64, H=1)
    k_f32_to_f16<<<(64 * 256 + 255) / 256, 256, 0, stream>>>(w3, w16, 64 * 256);
    k_transform<<<dim3(M / 64, 1), 128, 0, stream>>>(x16, w16, h32, hB, 256, 64);
    k_scores<<<(NB * NNODES + 255) / 256, 256, 0, stream>>>(h32, as3, ad3, sb, db, 1, 64);
    k_gat_attn<<<dim3(NNODES / 16, NB), 32, 0, stream>>>(adj, sb, db, hB, b3, x16, xn, 1, 64, 1);

    // ---- readout
    k_readout_g<<<NB, 64, 0, stream>>>(xn, wg, bg, wv, pv);
    k_final<<<(M + 255) / 256, 256, 0, stream>>>(xn, wn, bn, wv, bv, pv, (float*)d_out);
}
